// SelfAttention_35519379538075
// MI455X (gfx1250) — matrix operand order repaired and checked by reading, not compiled, not run on hardware
//
#include <hip/hip_runtime.h>
#include <hip/hip_bf16.h>
#include <math.h>

// ---------------------------------------------------------------------------
// Self-attention, CDNA5 (gfx1250) WMMA implementation.
// B=2 L=2048 D=1024 H=16 HD=64.  All GEMMs: v_wmma_f32_16x16x32_f16.
// Data movement: GLOBAL_LOAD_ASYNC_TO_LDS_B128 (ASYNCcnt) for pure-f16 tile
// copies, manual staging only where a transpose or f32->f16 convert is needed.
// ---------------------------------------------------------------------------

typedef _Float16 h8   __attribute__((ext_vector_type(8)));
typedef _Float16 v16h __attribute__((ext_vector_type(16)));
typedef float    v8f  __attribute__((ext_vector_type(8)));

static constexpr int   kB  = 2;
static constexpr int   kL  = 2048;
static constexpr int   kD  = 1024;
static constexpr int   kH  = 16;
static constexpr int   kHD = 64;
static constexpr float kScale = 0.125f;   // 1/sqrt(64)

// ---- async global->LDS copy (CDNA5 async engine, ISA ch.8) -----------------
// LDS generic pointers are {SHARED_BASE, offset[31:0]} (ISA 10.2), so the low
// 32 bits of a generic pointer to __shared__ are the LDS byte offset.
__device__ __forceinline__ unsigned lds_off(const void* p) {
  return (unsigned)(unsigned long long)p;
}

// one 16-byte chunk, per-lane addresses; tracked by ASYNCcnt
__device__ __forceinline__ void async_copy_b128(unsigned lds_byte,
                                                const void* gsrc) {
  asm volatile("global_load_async_to_lds_b128 %0, %1, off"
               :: "v"(lds_byte), "v"(gsrc)
               : "memory");
}

__device__ __forceinline__ void wait_async0() {
#if __has_builtin(__builtin_amdgcn_s_wait_asynccnt)
  __builtin_amdgcn_s_wait_asynccnt(0);
#else
  asm volatile("s_wait_asynccnt 0" ::: "memory");
#endif
}

// ---- WMMA fragment helpers (layouts per CDNA5 ISA 7.12.2, wave32) ----------
//
// A (16x32 f16):  lane m = lane&15, kb = (lane>>4)*8
//   a[j]   = A[m][kb+j]       j=0..7
//   a[8+j] = A[m][kb+16+j]    j=0..7
// B (32x16 f16) read from a source indexed [n][k]:  n = lane&15, kb = (lane>>4)*8
//   b[j]   = Bt[n][kb+j]      j=0..7
//   b[8+j] = Bt[n][kb+16+j]   j=0..7
// C/D (16x16 f32): c[r] = C[r + 8*(lane>>4)][lane&15]

__device__ __forceinline__ v16h frag_a(const _Float16* base, int stride, int lane) {
  const int m  = lane & 15;
  const int kb = (lane >> 4) << 3;
  const _Float16* p = base + m * stride + kb;
  h8 lo = *(const h8*)(p);
  h8 hi = *(const h8*)(p + 16);
  return __builtin_shufflevector(lo, hi, 0,1,2,3,4,5,6,7,8,9,10,11,12,13,14,15);
}

__device__ __forceinline__ v16h frag_b(const _Float16* base, int stride, int lane) {
  const int n  = lane & 15;
  const int kb = (lane >> 4) << 3;
  const _Float16* p = base + n * stride + kb;
  h8 lo = *(const h8*)(p);
  h8 hi = *(const h8*)(p + 16);
  return __builtin_shufflevector(lo, hi, 0,1,2,3,4,5,6,7,8,9,10,11,12,13,14,15);
}

__device__ __forceinline__ v8f wmma(v16h a, v16h b, v8f c) {
  return __builtin_amdgcn_wmma_f32_16x16x32_f16(false, a, false, b, (short)0, c,
                                                false, false);
}

// ---------------------------------------------------------------------------
// Kernel 1: QKV projection + RoPE.
// grid (48, 64), block 128 (4 waves).  blockIdx.x = g = sel*16 + h selects 64
// consecutive columns of W_qkv; blockIdx.y selects 64 rows of x (B*L major).
// Each wave computes a 16x64 strip (4 accumulator tiles).
// Staging needs f32->f16 conversion, so it stays through VGPRs; next weight
// tile is prefetched (global_prefetch_b8).
// ---------------------------------------------------------------------------
__global__ __launch_bounds__(128) void qkv_rope_kernel(
    const float* __restrict__ x, const float* __restrict__ cosb,
    const float* __restrict__ sinb, const float* __restrict__ Wqkv,
    _Float16* __restrict__ qws, _Float16* __restrict__ kws,
    _Float16* __restrict__ vws)
{
  __shared__ __align__(16) _Float16 Ah[64 * 40];
  __shared__ __align__(16) _Float16 Bt[64 * 40];

  const int tid  = threadIdx.x;
  const int lane = tid & 31;
  const int wv   = tid >> 5;
  const int g    = blockIdx.x;          // sel*16 + h
  const int m0   = blockIdx.y * 64;     // row base into (B*L)
  const int colb = g * 64;              // column base into 3*D

  v8f acc[4] = {{}, {}, {}, {}};

  for (int k0 = 0; k0 < kD; k0 += 32) {
    // prefetch next K-step tiles while this one is staged/consumed
    if (k0 + 32 < kD) {
      __builtin_prefetch(&Wqkv[(k0 + 32 + (tid >> 2)) * (3 * kD) + colb +
                               ((tid & 3) << 4)], 0, 1);
      __builtin_prefetch(&x[(m0 + (tid >> 1)) * kD + k0 + 32 +
                            ((tid & 1) << 4)], 0, 1);
    }
    // stage A tile 64x32 (f32 -> f16)
    #pragma unroll
    for (int i = 0; i < 16; ++i) {
      int e = tid + 128 * i;
      int r = e >> 5, c = e & 31;
      Ah[r * 40 + c] = (_Float16)x[(m0 + r) * kD + k0 + c];
    }
    // stage B tile 32x64 transposed: Bt[n][k]
    #pragma unroll
    for (int i = 0; i < 16; ++i) {
      int e = tid + 128 * i;
      int kk = e >> 6, nn = e & 63;
      Bt[nn * 40 + kk] = (_Float16)Wqkv[(k0 + kk) * (3 * kD) + colb + nn];
    }
    __syncthreads();

    v16h a = frag_a(Ah + wv * 16 * 40, 40, lane);
    #pragma unroll
    for (int t = 0; t < 4; ++t) {
      v16h b = frag_b(Bt + t * 16 * 40, 40, lane);
      acc[t] = wmma(a, b, acc[t]);
    }
    __syncthreads();
  }

  const int n  = lane & 15;
  const int rh = (lane >> 4) * 8;
  const int sel = g >> 4, h = g & 15;

  if (sel == 2) {                       // V: plain store, head-major layout
    #pragma unroll
    for (int t = 0; t < 4; ++t)
      #pragma unroll
      for (int r = 0; r < 8; ++r) {
        int R = m0 + wv * 16 + rh + r;
        int b = R >> 11, l = R & (kL - 1);
        vws[(((long)(b * kH + h) * kL) + l) * kHD + t * 16 + n] =
            (_Float16)acc[t][r];
      }
  } else {                              // Q/K: fused RoPE (pairs d, d+32)
    _Float16* dst = (sel == 0) ? qws : kws;
    #pragma unroll
    for (int t = 0; t < 2; ++t)
      #pragma unroll
      for (int r = 0; r < 8; ++r) {
        int R = m0 + wv * 16 + rh + r;
        int b = R >> 11, l = R & (kL - 1);
        int d = t * 16 + n;             // 0..31
        int ci = (b * kL + l) * (kD / 2) + h * 32 + d;
        float c  = cosb[ci];
        float s  = sinb[ci];
        float x1 = acc[t][r];
        float x2 = acc[t + 2][r];
        long base = (((long)(b * kH + h) * kL) + l) * kHD;
        dst[base + d]      = (_Float16)(x1 * c - x2 * s);
        dst[base + d + 32] = (_Float16)(x1 * s + x2 * c);
      }
  }
}

// ---------------------------------------------------------------------------
// Kernel 2: flash attention.  grid (L/64, B*H), block 128 (4 waves).
// Wave handles 16 q rows x HD=64.  Key loop tiles of 64.
// K tile: async DMA global->LDS (pure f16 copy, read once per block instead
// of once per wave).  V tile: manual transposed staging (DMA cannot
// transpose).  P round-trips through per-wave LDS to become A-fragments.
// ---------------------------------------------------------------------------
__global__ __launch_bounds__(128) void attn_kernel(
    const _Float16* __restrict__ qws, const _Float16* __restrict__ kws,
    const _Float16* __restrict__ vws, const int* __restrict__ mask,
    _Float16* __restrict__ aout)
{
  __shared__ __align__(16) _Float16 Kt[64 * 72];       // K tile, [key][hd]
  __shared__ __align__(16) _Float16 Vt[64 * 72];       // V tile, [hd][key]
  __shared__ __align__(16) _Float16 Pl[4 * 16 * 72];   // per-wave P staging

  const int tid  = threadIdx.x;
  const int lane = tid & 31;
  const int wv   = tid >> 5;
  const int bh   = blockIdx.y;          // b*16 + h
  const int b    = bh >> 4;
  const int h    = bh & 15;
  const int q0   = blockIdx.x * 64 + wv * 16;
  const int n    = lane & 15;
  const int rh   = (lane >> 4) * 8;

  const _Float16* qh = qws + (long)bh * kL * kHD;
  const _Float16* kh = kws + (long)bh * kL * kHD;
  const _Float16* vh = vws + (long)bh * kL * kHD;

  // Q fragments (persist across the whole key loop)
  v16h qa0 = frag_a(qh + q0 * kHD,      kHD, lane);
  v16h qa1 = frag_a(qh + q0 * kHD + 32, kHD, lane);

  v8f o[4] = {{}, {}, {}, {}};
  float M[8], Ls[8];
  #pragma unroll
  for (int r = 0; r < 8; ++r) { M[r] = -1e30f; Ls[r] = 0.f; }

  for (int kt = 0; kt < kL; kt += 64) {
    __syncthreads();
    // ---- K tile: async DMA, 64 rows x 128B = 512 x 16B chunks ----
    #pragma unroll
    for (int i = 0; i < 4; ++i) {
      int u  = tid + 128 * i;           // 0..511
      int r  = u >> 3;                  // key row 0..63
      int c8 = (u & 7) * 8;             // half-offset chunk
      async_copy_b128(lds_off(&Kt[r * 72 + c8]),
                      kh + (kt + r) * kHD + c8);
    }
    // ---- V tile: manual transposed staging Vt[hd][key] ----
    #pragma unroll
    for (int i = 0; i < 32; ++i) {
      int e = tid + 128 * i;
      int key = e >> 6, hd = e & 63;
      Vt[hd * 72 + key] = vh[(kt + key) * kHD + hd];
    }
    wait_async0();
    __syncthreads();

    // ---- S = Q K^T ----
    v8f s[4] = {{}, {}, {}, {}};
    #pragma unroll
    for (int t = 0; t < 4; ++t) {
      v16h bk0 = frag_b(Kt + t * 16 * 72,      72, lane);
      v16h bk1 = frag_b(Kt + t * 16 * 72 + 32, 72, lane);
      s[t] = wmma(qa0, bk0, s[t]);
      s[t] = wmma(qa1, bk1, s[t]);
    }

    // mask bias (per key column of each tile)
    float mb[4];
    #pragma unroll
    for (int t = 0; t < 4; ++t)
      mb[t] = (mask[b * kL + kt + t * 16 + n] == 0) ? -1e30f : 0.f;

    // ---- online softmax ----
    float mnew[8];
    #pragma unroll
    for (int r = 0; r < 8; ++r) {
      float mt = M[r];
      #pragma unroll
      for (int t = 0; t < 4; ++t) {
        float sv = s[t][r] * kScale + mb[t];
        s[t][r] = sv;
        mt = fmaxf(mt, sv);
      }
      #pragma unroll
      for (int off = 1; off < 16; off <<= 1)
        mt = fmaxf(mt, __shfl_xor(mt, off, 32));
      mnew[r] = mt;
    }
    #pragma unroll
    for (int r = 0; r < 8; ++r) {
      float corr = __expf(M[r] - mnew[r]);
      float ls = 0.f;
      #pragma unroll
      for (int t = 0; t < 4; ++t) {
        float p = __expf(s[t][r] - mnew[r]);
        s[t][r] = p;
        ls += p;
      }
      #pragma unroll
      for (int off = 1; off < 16; off <<= 1)
        ls += __shfl_xor(ls, off, 32);
      Ls[r] = Ls[r] * corr + ls;
      M[r]  = mnew[r];
      #pragma unroll
      for (int t = 0; t < 4; ++t)
        o[t][r] *= corr;
    }

    // ---- P (C-layout) -> per-wave LDS -> A fragments ----
    _Float16* pw = Pl + wv * 16 * 72;
    #pragma unroll
    for (int t = 0; t < 4; ++t)
      #pragma unroll
      for (int r = 0; r < 8; ++r)
        pw[(rh + r) * 72 + t * 16 + n] = (_Float16)s[t][r];
    // same-wave DS RAW: compiler inserts s_wait_dscnt; region is per-wave.
    v16h pa0 = frag_a(pw,      72, lane);
    v16h pa1 = frag_a(pw + 32, 72, lane);

    // ---- O += P V ----
    #pragma unroll
    for (int t = 0; t < 4; ++t) {
      v16h bv0 = frag_b(Vt + t * 16 * 72,      72, lane);
      v16h bv1 = frag_b(Vt + t * 16 * 72 + 32, 72, lane);
      o[t] = wmma(pa0, bv0, o[t]);
      o[t] = wmma(pa1, bv1, o[t]);
    }
  }

  // normalize and store to (B*L, D) f16, col = h*64 + d
  #pragma unroll
  for (int r = 0; r < 8; ++r) {
    float inv = 1.f / Ls[r];
    int l = q0 + rh + r;
    #pragma unroll
    for (int t = 0; t < 4; ++t)
      aout[((long)(b * kL + l)) * kD + h * kHD + t * 16 + n] =
          (_Float16)(o[t][r] * inv);
  }
}

// ---------------------------------------------------------------------------
// Kernel 3: output projection attn(4096x1024 f16) @ W_o(1024x1024) -> f32.
// grid (16, 64), block 128.  A tile is already f16: async DMA to LDS.
// ---------------------------------------------------------------------------
__global__ __launch_bounds__(128) void out_proj_kernel(
    const _Float16* __restrict__ A, const float* __restrict__ Wo,
    float* __restrict__ out)
{
  __shared__ __align__(16) _Float16 Ah[64 * 40];
  __shared__ __align__(16) _Float16 Bt[64 * 40];

  const int tid  = threadIdx.x;
  const int lane = tid & 31;
  const int wv   = tid >> 5;
  const int m0   = blockIdx.y * 64;
  const int colb = blockIdx.x * 64;

  v8f acc[4] = {{}, {}, {}, {}};

  for (int k0 = 0; k0 < kD; k0 += 32) {
    if (k0 + 32 < kD)
      __builtin_prefetch(&Wo[(k0 + 32 + (tid >> 2)) * kD + colb +
                             ((tid & 3) << 4)], 0, 1);
    // ---- A tile 64x32 f16: async DMA, 256 x 16B chunks ----
    #pragma unroll
    for (int i = 0; i < 2; ++i) {
      int u  = tid + 128 * i;           // 0..255
      int r  = u >> 2;                  // row 0..63
      int c8 = (u & 3) * 8;             // half-offset chunk
      async_copy_b128(lds_off(&Ah[r * 40 + c8]),
                      A + (long)(m0 + r) * kD + k0 + c8);
    }
    // ---- B tile 32x64 transposed with f32->f16 convert ----
    #pragma unroll
    for (int i = 0; i < 16; ++i) {
      int e = tid + 128 * i;
      int kk = e >> 6, nn = e & 63;
      Bt[nn * 40 + kk] = (_Float16)Wo[(k0 + kk) * kD + colb + nn];
    }
    wait_async0();
    __syncthreads();

    v16h a = frag_a(Ah + wv * 16 * 40, 40, lane);
    #pragma unroll
    for (int t = 0; t < 4; ++t) {
      v16h b = frag_b(Bt + t * 16 * 40, 40, lane);
      acc[t] = wmma(a, b, acc[t]);
    }
    __syncthreads();
  }

  const int n = lane & 15, rh = (lane >> 4) * 8;
  #pragma unroll
  for (int t = 0; t < 4; ++t)
    #pragma unroll
    for (int r = 0; r < 8; ++r)
      out[(long)(m0 + wv * 16 + rh + r) * kD + colb + t * 16 + n] = acc[t][r];
}

// ---------------------------------------------------------------------------
extern "C" void kernel_launch(void* const* d_in, const int* in_sizes, int n_in,
                              void* d_out, int out_size, void* d_ws,
                              size_t ws_size, hipStream_t stream) {
  const float* x    = (const float*)d_in[0];
  const float* rc   = (const float*)d_in[1];
  const float* rs   = (const float*)d_in[2];
  const float* wqkv = (const float*)d_in[3];
  const float* wo   = (const float*)d_in[4];
  const int*   mask = (const int*)d_in[5];

  const long perTensor = (long)kB * kH * kL * kHD;   // 4,194,304 halfs
  _Float16* ws  = (_Float16*)d_ws;
  _Float16* qws = ws;
  _Float16* kws = ws + perTensor;
  _Float16* vws = ws + 2 * perTensor;
  _Float16* aws = ws + 3 * perTensor;

  dim3 blk(128);
  qkv_rope_kernel<<<dim3(3 * kH, (kB * kL) / 64), blk, 0, stream>>>(
      x, rc, rs, wqkv, qws, kws, vws);
  attn_kernel<<<dim3(kL / 64, kB * kH), blk, 0, stream>>>(
      qws, kws, vws, mask, aws);
  out_proj_kernel<<<dim3(kD / 64, (kB * kL) / 64), blk, 0, stream>>>(
      aws, wo, (float*)d_out);
}
